// MultiHeadAttention_42417097015515
// MI455X (gfx1250) — compile-verified, matrix operand order checked by reading, not run on hardware
//
#include <hip/hip_runtime.h>
#include <hip/hip_bf16.h>

// ---------------------------------------------------------------------------
// Multi-head attention for MI455X (gfx1250), bf16 WMMA pipeline.
// B=4, S=2048, D=1024, H=16, HD=64.
// Round 2: 32-row wave tiles everywhere (A/B fragment reuse -> load:wmma 1.5)
// ---------------------------------------------------------------------------

#define BATCH 4
#define SEQ   2048
#define DIM   1024
#define HEADS 16
#define HDIM  64
#define NQKV  (3 * DIM)     // 3072

typedef __attribute__((ext_vector_type(16))) __bf16       v16bf;
typedef __attribute__((ext_vector_type(8)))  float        v8f;
typedef __attribute__((ext_vector_type(4)))  unsigned int v4u;

union FragBF {
    v16bf v;
    v4u   u[2];
};

// fp32 -> bf16 round-to-nearest-even
__device__ __forceinline__ unsigned short f2bf(float x) {
    unsigned int u = __float_as_uint(x);
    unsigned int r = u + 0x7FFFu + ((u >> 16) & 1u);
    return (unsigned short)(r >> 16);
}

// Unified 16x32 (A) / 32x16 (B) bf16 fragment loader from row-major storage.
// For an A tile: row = M index of this lane (lane%16).
// For a B tile (B[k][n] = Mrow[n][k]): row = N index of this lane (lane%16).
// K striping per ISA 7.12.2: half h = lane>>4 holds K in
// [kbase+8h, kbase+8h+7] and [kbase+16+8h, kbase+23+8h]  -> two b128 loads.
__device__ __forceinline__ FragBF load_frag(const unsigned short* base,
                                            int row, int ld, int kbase, int h) {
    FragBF f;
    const unsigned short* p = base + (size_t)row * ld + kbase + 8 * h;
    f.u[0] = *reinterpret_cast<const v4u*>(p);
    f.u[1] = *reinterpret_cast<const v4u*>(p + 16);
    return f;
}

__device__ __forceinline__ v8f wmma_bf16(const FragBF& a, const FragBF& b, v8f c) {
    return __builtin_amdgcn_wmma_f32_16x16x32_bf16(false, a.v, false, b.v,
                                                   (short)0, c, false, false);
}

// ---------------------------------------------------------------------------
// Kernel 0: fp32 -> bf16 conversion (grid-stride)
// ---------------------------------------------------------------------------
__global__ void cvt_bf16_kernel(const float* __restrict__ in,
                                unsigned short* __restrict__ out, int n) {
    int i = blockIdx.x * blockDim.x + threadIdx.x;
    int stride = gridDim.x * blockDim.x;
    for (; i < n; i += stride) out[i] = f2bf(in[i]);
}

// ---------------------------------------------------------------------------
// Kernel 1: QKV projection.  qkv[m][n] = sum_k x[m][k] * Wqkv[n][k] + b[n]
// M = B*S = 8192, N = 3072, K = 1024.
// Wave tile: 32 x 64 (2 A frags x 4 B frags = 8 WMMAs / k-step).
// Block: 8 waves -> 256 x 64.  Grid (48, 32).
// Scatter: n -> head = n/192, c = n%192; c<64 -> Q[b,h,s,c] ;
// c<128 -> K[b,h,s,c-64] ; else V^T[b,h,c-128,s] (b128 store).
// ---------------------------------------------------------------------------
__global__ __launch_bounds__(256) void qkv_gemm_kernel(
    const unsigned short* __restrict__ xb,
    const unsigned short* __restrict__ wq,
    const float* __restrict__ bias,
    unsigned short* __restrict__ Qb,
    unsigned short* __restrict__ Kb,
    unsigned short* __restrict__ Vt) {
    const int tid  = threadIdx.x;
    const int wave = tid >> 5;
    const int lane = tid & 31;
    const int hf   = lane >> 4;
    const int lr   = lane & 15;
    const int m0   = blockIdx.y * 256 + wave * 32;
    const int n0   = blockIdx.x * 64;

    v8f acc[8] = {};
    for (int k = 0; k < DIM; k += 32) {
        if (k + 32 < DIM) {
            __builtin_prefetch(xb + (size_t)(m0 + lr) * DIM + k + 32, 0, 1);
            __builtin_prefetch(xb + (size_t)(m0 + 16 + lr) * DIM + k + 32, 0, 1);
        }
        FragBF a0 = load_frag(xb, m0 + lr, DIM, k, hf);
        FragBF a1 = load_frag(xb, m0 + 16 + lr, DIM, k, hf);
#pragma unroll
        for (int j = 0; j < 4; ++j) {
            FragBF b = load_frag(wq, n0 + j * 16 + lr, DIM, k, hf);
            acc[j]     = wmma_bf16(a0, b, acc[j]);
            acc[4 + j] = wmma_bf16(a1, b, acc[4 + j]);
        }
    }

    const int bb = m0 / SEQ;   // tiles never straddle a batch boundary
#pragma unroll
    for (int t = 0; t < 2; ++t) {
        const int s_base = (m0 % SEQ) + t * 16;
#pragma unroll
        for (int j = 0; j < 4; ++j) {
            const v8f& a   = acc[t * 4 + j];
            const int n    = n0 + j * 16 + lr;
            const int head = n / (3 * HDIM);
            const int c    = n % (3 * HDIM);
            const float bv = bias[n];
            if (c < 2 * HDIM) {
                unsigned short* dst = (c < HDIM) ? Qb : Kb;
                const int cc = c & (HDIM - 1);
                const size_t base = (((size_t)bb * HEADS + head) * SEQ) * HDIM + cc;
#pragma unroll
                for (int r = 0; r < 8; ++r) {
                    const int s = s_base + hf * 8 + r;
                    dst[base + (size_t)s * HDIM] = f2bf(a[r] + bv);
                }
            } else {
                const int cc = c - 2 * HDIM;
                const size_t base = (((size_t)bb * HEADS + head) * HDIM + cc) * SEQ;
                alignas(16) unsigned short pack[8];
#pragma unroll
                for (int r = 0; r < 8; ++r) pack[r] = f2bf(a[r] + bv);
                *reinterpret_cast<v4u*>(&Vt[base + s_base + hf * 8]) =
                    *reinterpret_cast<const v4u*>(pack);
            }
        }
    }
}

// ---------------------------------------------------------------------------
// Kernel 2: flash attention.  One wave = one 32-query tile of one (b,h),
// as two 16-row sub-tiles sharing every K- and V-fragment.
// Per 32-key block: 8 score WMMAs + online softmax + LDS transpose of P
// + 8 PV WMMAs  (16 WMMAs per 16 b128 K/V loads).
// ---------------------------------------------------------------------------
__global__ __launch_bounds__(256) void attn_kernel(
    const unsigned short* __restrict__ Qb,
    const unsigned short* __restrict__ Kb,
    const unsigned short* __restrict__ Vt,
    unsigned short* __restrict__ Ob) {
    __shared__ unsigned short lds_p[8][32 * 32];  // per-wave P tile (2 KB)

    const int tid  = threadIdx.x;
    const int wave = tid >> 5;
    const int lane = tid & 31;
    const int hf   = lane >> 4;
    const int lr   = lane & 15;
    const int h    = blockIdx.y;
    const int b    = blockIdx.z;
    const int s0   = (blockIdx.x * 8 + wave) * 32;

    const unsigned short* Qh = Qb + (((size_t)b * HEADS + h) * SEQ) * HDIM;
    const unsigned short* Kh = Kb + (((size_t)b * HEADS + h) * SEQ) * HDIM;
    const unsigned short* Vh = Vt + (((size_t)b * HEADS + h) * HDIM) * SEQ;

    FragBF aq[2][2];
#pragma unroll
    for (int t = 0; t < 2; ++t) {
        aq[t][0] = load_frag(Qh, s0 + t * 16 + lr, HDIM, 0, hf);
        aq[t][1] = load_frag(Qh, s0 + t * 16 + lr, HDIM, 32, hf);
    }

    v8f o[2][4] = {};
    float mrow[2][8], lrow[2][8];
#pragma unroll
    for (int t = 0; t < 2; ++t)
#pragma unroll
        for (int r = 0; r < 8; ++r) { mrow[t][r] = -1.0e30f; lrow[t][r] = 0.0f; }
    const float scale = 0.125f;  // 1/sqrt(64)

    unsigned short* lp = &lds_p[wave][0];

    for (int kb = 0; kb < SEQ; kb += 32) {
        // K fragments for this 32-key block (shared by both query sub-tiles)
        FragBF bk[2][2];
#pragma unroll
        for (int kj = 0; kj < 2; ++kj) {
            bk[kj][0] = load_frag(Kh, kb + kj * 16 + lr, HDIM, 0, hf);
            bk[kj][1] = load_frag(Kh, kb + kj * 16 + lr, HDIM, 32, hf);
        }
        // scores: 32 queries x 32 keys, f32
        v8f sc[2][2];
#pragma unroll
        for (int t = 0; t < 2; ++t)
#pragma unroll
            for (int kj = 0; kj < 2; ++kj) {
                v8f s = {};
                s = wmma_bf16(aq[t][0], bk[kj][0], s);
                s = wmma_bf16(aq[t][1], bk[kj][1], s);
                sc[t][kj] = s;
            }
        // online softmax per score row (row r+8*hf of sub-tile t)
#pragma unroll
        for (int t = 0; t < 2; ++t) {
#pragma unroll
            for (int r = 0; r < 8; ++r) {
                float x0 = sc[t][0][r] * scale;
                float x1 = sc[t][1][r] * scale;
                float mx = fmaxf(x0, x1);
                mx = fmaxf(mx, __shfl_xor(mx, 1, 32));
                mx = fmaxf(mx, __shfl_xor(mx, 2, 32));
                mx = fmaxf(mx, __shfl_xor(mx, 4, 32));
                mx = fmaxf(mx, __shfl_xor(mx, 8, 32));
                const float mnew  = fmaxf(mrow[t][r], mx);
                const float alpha = __expf(mrow[t][r] - mnew);
                const float p0 = __expf(x0 - mnew);
                const float p1 = __expf(x1 - mnew);
                float ps = p0 + p1;
                ps += __shfl_xor(ps, 1, 32);
                ps += __shfl_xor(ps, 2, 32);
                ps += __shfl_xor(ps, 4, 32);
                ps += __shfl_xor(ps, 8, 32);
                lrow[t][r] = lrow[t][r] * alpha + ps;
                mrow[t][r] = mnew;
                o[t][0][r] *= alpha; o[t][1][r] *= alpha;
                o[t][2][r] *= alpha; o[t][3][r] *= alpha;
                sc[t][0][r] = p0; sc[t][1][r] = p1;
            }
        }
        // C-layout -> A-layout transpose of P (32x32) through LDS
        __syncthreads();
#pragma unroll
        for (int t = 0; t < 2; ++t)
#pragma unroll
            for (int r = 0; r < 8; ++r) {
                const int row = t * 16 + hf * 8 + r;
                lp[row * 32 + lr]      = f2bf(sc[t][0][r]);
                lp[row * 32 + 16 + lr] = f2bf(sc[t][1][r]);
            }
        __syncthreads();
        FragBF ap[2];
        ap[0] = load_frag(lp, lr, 32, 0, hf);
        ap[1] = load_frag(lp, 16 + lr, 32, 0, hf);
        // O += P * V   (V^T rows contiguous over keys -> b128 loads, shared)
#pragma unroll
        for (int j = 0; j < 4; ++j) {
            FragBF bv = load_frag(Vh, j * 16 + lr, SEQ, kb, hf);
            o[0][j] = wmma_bf16(ap[0], bv, o[0][j]);
            o[1][j] = wmma_bf16(ap[1], bv, o[1][j]);
        }
    }

    // normalize + store bf16 O in [B, S, D] layout
#pragma unroll
    for (int t = 0; t < 2; ++t)
#pragma unroll
        for (int r = 0; r < 8; ++r) {
            const float inv = 1.0f / lrow[t][r];
            const int s = s0 + t * 16 + hf * 8 + r;
            const size_t base = ((size_t)b * SEQ + s) * DIM + h * HDIM;
#pragma unroll
            for (int j = 0; j < 4; ++j)
                Ob[base + j * 16 + lr] = f2bf(o[t][j][r] * inv);
        }
}

// ---------------------------------------------------------------------------
// Kernel 3: output projection.  out[m][n] = sum_k O[m][k]*Wout[n][k] + b[n]
// M = 8192, N = 1024, K = 1024.  Wave tile 32 x 64; f32 output.
// ---------------------------------------------------------------------------
__global__ __launch_bounds__(256) void out_gemm_kernel(
    const unsigned short* __restrict__ Ob,
    const unsigned short* __restrict__ wo,
    const float* __restrict__ bias,
    float* __restrict__ out) {
    const int tid  = threadIdx.x;
    const int wave = tid >> 5;
    const int lane = tid & 31;
    const int hf   = lane >> 4;
    const int lr   = lane & 15;
    const int m0   = blockIdx.y * 256 + wave * 32;
    const int n0   = blockIdx.x * 64;

    v8f acc[8] = {};
    for (int k = 0; k < DIM; k += 32) {
        if (k + 32 < DIM) {
            __builtin_prefetch(Ob + (size_t)(m0 + lr) * DIM + k + 32, 0, 1);
            __builtin_prefetch(Ob + (size_t)(m0 + 16 + lr) * DIM + k + 32, 0, 1);
        }
        FragBF a0 = load_frag(Ob, m0 + lr, DIM, k, hf);
        FragBF a1 = load_frag(Ob, m0 + 16 + lr, DIM, k, hf);
#pragma unroll
        for (int j = 0; j < 4; ++j) {
            FragBF b = load_frag(wo, n0 + j * 16 + lr, DIM, k, hf);
            acc[j]     = wmma_bf16(a0, b, acc[j]);
            acc[4 + j] = wmma_bf16(a1, b, acc[4 + j]);
        }
    }
#pragma unroll
    for (int t = 0; t < 2; ++t)
#pragma unroll
        for (int j = 0; j < 4; ++j) {
            const int n    = n0 + j * 16 + lr;
            const float bv = bias[n];
#pragma unroll
            for (int r = 0; r < 8; ++r) {
                const int m = m0 + t * 16 + hf * 8 + r;
                out[(size_t)m * DIM + n] = acc[t * 4 + j][r] + bv;
            }
        }
}

// ---------------------------------------------------------------------------
// Workspace layout (bf16 elements)
// ---------------------------------------------------------------------------
static constexpr size_t XB_N   = (size_t)BATCH * SEQ * DIM;   // 8388608
static constexpr size_t WQKV_N = (size_t)NQKV * DIM;          // 3145728
static constexpr size_t WOUT_N = (size_t)DIM * DIM;           // 1048576
static constexpr size_t OFF_XB   = 0;
static constexpr size_t OFF_WQKV = OFF_XB + XB_N;
static constexpr size_t OFF_WOUT = OFF_WQKV + WQKV_N;
static constexpr size_t OFF_Q    = OFF_WOUT + WOUT_N;
static constexpr size_t OFF_K    = OFF_Q + XB_N;
static constexpr size_t OFF_V    = OFF_K + XB_N;
static constexpr size_t OFF_O    = OFF_V + XB_N;

extern "C" void kernel_launch(void* const* d_in, const int* in_sizes, int n_in,
                              void* d_out, int out_size, void* d_ws, size_t ws_size,
                              hipStream_t stream) {
    const float* x     = (const float*)d_in[0];
    const float* W_qkv = (const float*)d_in[1];
    const float* b_qkv = (const float*)d_in[2];
    const float* W_out = (const float*)d_in[3];
    const float* b_out = (const float*)d_in[4];
    float* out = (float*)d_out;

    unsigned short* ws = (unsigned short*)d_ws;
    unsigned short* xb   = ws + OFF_XB;
    unsigned short* wqkv = ws + OFF_WQKV;
    unsigned short* wout = ws + OFF_WOUT;
    unsigned short* Qb   = ws + OFF_Q;
    unsigned short* Kb   = ws + OFF_K;
    unsigned short* Vt   = ws + OFF_V;
    unsigned short* Ob   = ws + OFF_O;

    // 0) fp32 -> bf16 conversions
    cvt_bf16_kernel<<<2048, 256, 0, stream>>>(x, xb, (int)XB_N);
    cvt_bf16_kernel<<<1024, 256, 0, stream>>>(W_qkv, wqkv, (int)WQKV_N);
    cvt_bf16_kernel<<<512, 256, 0, stream>>>(W_out, wout, (int)WOUT_N);

    // 1) fused QKV projection -> Q[B,H,S,HD], K[B,H,S,HD], V^T[B,H,HD,S]
    qkv_gemm_kernel<<<dim3(NQKV / 64, (BATCH * SEQ) / 256), 256, 0, stream>>>(
        xb, wqkv, b_qkv, Qb, Kb, Vt);

    // 2) flash attention -> O bf16 in [B, S, D]
    attn_kernel<<<dim3(SEQ / 256, HEADS, BATCH), 256, 0, stream>>>(Qb, Kb, Vt, Ob);

    // 3) output projection -> f32 out [B, S, D]
    out_gemm_kernel<<<dim3(DIM / 64, (BATCH * SEQ) / 256), 256, 0, stream>>>(
        Ob, wout, b_out, out);
}
